// LSTM_RNN_69793218560220
// MI455X (gfx1250) — matrix operand order repaired and checked by reading, not compiled, not run on hardware
//
#include <hip/hip_runtime.h>

#define T_SEQ  512
#define BATCH  64
#define EMBD   512
#define UNITS  512
#define NCLS   10

typedef _Float16 v16h __attribute__((ext_vector_type(16)));
typedef _Float16 v8h  __attribute__((ext_vector_type(8)));
typedef float    v8f  __attribute__((ext_vector_type(8)));

// ---------------------------------------------------------------------------
// Fragment loaders.
//  A (16-bit A 16x32 layout): halves[0..7] from p, halves[8..15] from p+16.
//  B (16-bit B 32x16 layout from N-major): halves[0..7] from p, halves[8..15] from p+16.
// ---------------------------------------------------------------------------
__device__ __forceinline__ v16h ldA(const _Float16* p)
{
    v8h lo = *(const v8h*)p;
    v8h hi = *(const v8h*)(p + 16);
    v16h a;
    #pragma unroll
    for (int i = 0; i < 8; ++i) { a[i] = lo[i]; a[i + 8] = hi[i]; }
    return a;
}
__device__ __forceinline__ v16h ldB(const _Float16* p)
{
    v8h lo = *(const v8h*)p;
    v8h hi = *(const v8h*)(p + 16);
    v16h b;
    #pragma unroll
    for (int i = 0; i < 8; ++i) { b[i] = lo[i]; b[i + 8] = hi[i]; }
    return b;
}

// ---------------------------------------------------------------------------
// 1) Embedding gather + cast to f16, layout (T, B, E).
// ---------------------------------------------------------------------------
__global__ void embed_gather_f16(const int* __restrict__ sent,
                                 const float* __restrict__ emb,
                                 _Float16* __restrict__ X)
{
    int row = blockIdx.x;              // row = t*BATCH + b
    int t = row >> 6;
    int b = row & 63;
    int tok = sent[b * T_SEQ + t];     // sentence is (B, T)
    const float* src = emb + (size_t)tok * EMBD;
    _Float16*    dst = X   + (size_t)row * EMBD;
    for (int e = threadIdx.x; e < EMBD; e += blockDim.x)
        dst[e] = (_Float16)src[e];
}

// ---------------------------------------------------------------------------
// 2) Cast f32 weights to f16 (row-major = N-major "B-transposed" layout).
// ---------------------------------------------------------------------------
__global__ void cast_f32_f16(const float* __restrict__ src,
                             _Float16* __restrict__ dst, int n)
{
    int i = blockIdx.x * blockDim.x + threadIdx.x;
    if (i < n) dst[i] = (_Float16)src[i];
}

// ---------------------------------------------------------------------------
// 3) Fused per-timestep LSTM step (one WG owns all 4 gates of a
//    64-batch x 16-unit block; software-pipelined WMMA K-loop; gate
//    pre-activations in LDS; fused sigmoid/tanh cell update).
// ---------------------------------------------------------------------------
__global__ __launch_bounds__(256) void lstm_step_fused(
    const _Float16* __restrict__ X,      // (T,B,E) f16
    const _Float16* __restrict__ Wf,     // (4,U,E) f16
    const _Float16* __restrict__ Uf,     // (4,U,U) f16
    const _Float16* __restrict__ hin,    // (B,U)   f16  (h_{t-1})
    _Float16* __restrict__ hout,         // (B,U)   f16  (h_t)
    float* __restrict__ c,               // (B,U)   f32  (in-place)
    float* __restrict__ h32,             // (B,U)   f32  (final h only)
    int t, int last)
{
    __shared__ float Gs[4][BATCH][16];   // 16 KB: all gates for this block

    const int lane  = threadIdx.x & 31;
    const int wave  = threadIdx.x >> 5;
    const int btile = wave & 3;          // batch tile 0..3
    const int g0    = (wave >> 2) << 1;  // gate pair: {0,1} or {2,3}
    const int g1    = g0 + 1;
    const int m0    = btile << 4;        // batch row base
    const int n0    = blockIdx.x << 4;   // unit col base (32 WGs)

    const int mrow  = m0 + (lane & 15);
    const int koffA = (lane >> 4) * 8;
    const int ncol  = n0 + (lane & 15);
    const int koffB = (lane >> 4) * 8;

    const _Float16* xa  = X   + ((size_t)t * BATCH + mrow) * EMBD  + koffA;
    const _Float16* ha  = hin +  (size_t)mrow * UNITS              + koffA;
    const _Float16* wb0 = Wf  + ((size_t)g0 * UNITS + ncol) * EMBD + koffB;
    const _Float16* wb1 = Wf  + ((size_t)g1 * UNITS + ncol) * EMBD + koffB;
    const _Float16* ub0 = Uf  + ((size_t)g0 * UNITS + ncol) * UNITS + koffB;
    const _Float16* ub1 = Uf  + ((size_t)g1 * UNITS + ncol) * UNITS + koffB;

    v8f accW0 = {}, accU0 = {}, accW1 = {}, accU1 = {};

    // Software-pipelined K-loop: prefetch chunk k+32 into fresh registers
    // before issuing chunk k's WMMAs, so WMMAs never wait on prefetches.
    v16h ax  = ldA(xa);
    v16h ah  = ldA(ha);
    v16h fw0 = ldB(wb0);
    v16h fw1 = ldB(wb1);
    v16h fu0 = ldB(ub0);
    v16h fu1 = ldB(ub1);

    #pragma unroll
    for (int k = 0; k < EMBD; k += 32) {
        v16h nax, nah, nw0, nw1, nu0, nu1;
        const int kn = k + 32;
        if (kn < EMBD) {
            nax = ldA(xa + kn);
            nah = ldA(ha + kn);
            nw0 = ldB(wb0 + kn);
            nw1 = ldB(wb1 + kn);
            nu0 = ldB(ub0 + kn);
            nu1 = ldB(ub1 + kn);
        }

        accW0 = __builtin_amdgcn_wmma_f32_16x16x32_f16(
                    false, ax, false, fw0, (short)0, accW0, false, false);
        accW1 = __builtin_amdgcn_wmma_f32_16x16x32_f16(
                    false, ax, false, fw1, (short)0, accW1, false, false);
        accU0 = __builtin_amdgcn_wmma_f32_16x16x32_f16(
                    false, ah, false, fu0, (short)0, accU0, false, false);
        accU1 = __builtin_amdgcn_wmma_f32_16x16x32_f16(
                    false, ah, false, fu1, (short)0, accU1, false, false);

        if (kn < EMBD) {
            ax = nax; ah = nah;
            fw0 = nw0; fw1 = nw1; fu0 = nu0; fu1 = nu1;
        }
    }

    // D layout: reg r, lane l -> row = m0 + r + 8*(l>>4), col = l&15
    const int brow = m0 + ((lane >> 4) << 3);
    const int ucol = lane & 15;
    #pragma unroll
    for (int r = 0; r < 8; ++r) {
        Gs[g0][brow + r][ucol] = accW0[r] + accU0[r];
        Gs[g1][brow + r][ucol] = accW1[r] + accU1[r];
    }
    __syncthreads();

    // Fused cell update for the (64 x 16) block this WG owns.
    for (int e = threadIdx.x; e < BATCH * 16; e += 256) {
        int b  = e >> 4;
        int ul = e & 15;
        size_t idx = (size_t)b * UNITS + (n0 + ul);
        float i_t = 1.0f / (1.0f + __expf(-Gs[0][b][ul]));
        float f_t = 1.0f / (1.0f + __expf(-Gs[1][b][ul]));
        float o_t = 1.0f / (1.0f + __expf(-Gs[2][b][ul]));
        float n_t = tanhf(Gs[3][b][ul]);
        float cn  = f_t * c[idx] + i_t * n_t;
        float hn  = o_t * tanhf(cn);
        c[idx]    = cn;
        hout[idx] = (_Float16)hn;
        if (last) h32[idx] = hn;
    }
}

// ---------------------------------------------------------------------------
// 4) MLP head + softmax (tiny: one workgroup).
// ---------------------------------------------------------------------------
__global__ __launch_bounds__(256) void head_softmax(
    const float* __restrict__ h,   // (B, UNITS)
    const float* __restrict__ W1,  // (UNITS, 32)
    const float* __restrict__ b1,  // (32)
    const float* __restrict__ W2,  // (32, NCLS)
    const float* __restrict__ b2,  // (NCLS)
    float* __restrict__ out)       // (B, NCLS)
{
    __shared__ float hid[BATCH][32];
    __shared__ float lg[BATCH][NCLS];

    for (int i = threadIdx.x; i < BATCH * 32; i += blockDim.x) {
        int b = i >> 5, j = i & 31;
        float s = b1[j];
        const float* hr = h + (size_t)b * UNITS;
        for (int k = 0; k < UNITS; ++k) s += hr[k] * W1[k * 32 + j];
        hid[b][j] = s > 0.0f ? s : 0.0f;
    }
    __syncthreads();

    for (int i = threadIdx.x; i < BATCH * NCLS; i += blockDim.x) {
        int b = i / NCLS, cc = i % NCLS;
        float s = b2[cc];
        for (int j = 0; j < 32; ++j) s += hid[b][j] * W2[j * NCLS + cc];
        lg[b][cc] = s;
    }
    __syncthreads();

    if (threadIdx.x < BATCH) {
        int b = threadIdx.x;
        float mx = lg[b][0];
        for (int cc = 1; cc < NCLS; ++cc) mx = fmaxf(mx, lg[b][cc]);
        float e[NCLS], sum = 0.0f;
        for (int cc = 0; cc < NCLS; ++cc) { e[cc] = __expf(lg[b][cc] - mx); sum += e[cc]; }
        float inv = 1.0f / sum;
        for (int cc = 0; cc < NCLS; ++cc) out[b * NCLS + cc] = e[cc] * inv;
    }
}

// ---------------------------------------------------------------------------
// Host orchestration. Inputs (setup_inputs order):
//   0: sentence (int32, B*T)   1: emb (f32, VOCAB*E)
//   2: W (f32, 4*H*E)          3: U (f32, 4*H*H)
//   4: W1 (f32, H*32)          5: b1 (f32, 32)
//   6: W2 (f32, 32*10)         7: b2 (f32, 10)
// Output: softmax (f32, B*10)
// ---------------------------------------------------------------------------
extern "C" void kernel_launch(void* const* d_in, const int* in_sizes, int n_in,
                              void* d_out, int out_size, void* d_ws, size_t ws_size,
                              hipStream_t stream)
{
    (void)in_sizes; (void)n_in; (void)out_size; (void)ws_size;

    const int*   sent = (const int*)d_in[0];
    const float* emb  = (const float*)d_in[1];
    const float* W    = (const float*)d_in[2];
    const float* U    = (const float*)d_in[3];
    const float* W1   = (const float*)d_in[4];
    const float* b1   = (const float*)d_in[5];
    const float* W2   = (const float*)d_in[6];
    const float* b2   = (const float*)d_in[7];
    float*       out  = (float*)d_out;

    // Workspace carve-up (256B aligned)
    char* base = (char*)d_ws;
    size_t off = 0;
    auto carve = [&](size_t bytes) -> void* {
        void* r = base + off;
        off += (bytes + 255) & ~(size_t)255;
        return r;
    };
    _Float16* Xf  = (_Float16*)carve((size_t)T_SEQ * BATCH * EMBD * sizeof(_Float16));
    _Float16* Wf  = (_Float16*)carve((size_t)4 * UNITS * EMBD * sizeof(_Float16));
    _Float16* Uf  = (_Float16*)carve((size_t)4 * UNITS * UNITS * sizeof(_Float16));
    float*    c   = (float*)carve((size_t)BATCH * UNITS * sizeof(float));
    float*    h32 = (float*)carve((size_t)BATCH * UNITS * sizeof(float));
    _Float16* hA  = (_Float16*)carve((size_t)BATCH * UNITS * sizeof(_Float16));
    _Float16* hB  = (_Float16*)carve((size_t)BATCH * UNITS * sizeof(_Float16));

    // Zero-init recurrent state (f16 zero is bit-pattern 0)
    hipMemsetAsync(c,  0, (size_t)BATCH * UNITS * sizeof(float),    stream);
    hipMemsetAsync(hA, 0, (size_t)BATCH * UNITS * sizeof(_Float16), stream);

    // One-time: embedding gather + weight casts
    embed_gather_f16<<<T_SEQ * BATCH, 256, 0, stream>>>(sent, emb, Xf);
    const int NW = 4 * UNITS * EMBD;   // == 4*UNITS*UNITS
    cast_f32_f16<<<(NW + 255) / 256, 256, 0, stream>>>(W, Wf, NW);
    cast_f32_f16<<<(NW + 255) / 256, 256, 0, stream>>>(U, Uf, NW);

    // Sequential scan: one fused kernel per timestep, h ping-pongs.
    _Float16* hbuf[2] = { hA, hB };
    for (int t = 0; t < T_SEQ; ++t) {
        lstm_step_fused<<<UNITS / 16, 256, 0, stream>>>(
            Xf, Wf, Uf, hbuf[t & 1], hbuf[(t + 1) & 1], c, h32,
            t, (t == T_SEQ - 1) ? 1 : 0);
    }

    // Head + softmax
    head_softmax<<<1, 256, 0, stream>>>(h32, W1, b1, W2, b2, out);
}
